// CuboidSelfAttention_40681930228070
// MI455X (gfx1250) — matrix operand order repaired and checked by reading, not compiled, not run on hardware
//
#include <hip/hip_runtime.h>

// ---------------------------------------------------------------------------
// CuboidSelfAttention for MI455X (gfx1250, wave32, WMMA).
// All matmuls use v_wmma_f32_16x16x32_f16 (f16 A/B, f32 accum).
// Memory-bound problem (~73 GFLOP vs ~1GB traffic @23.3TB/s) -> f16
// intermediates everywhere, fused LN/roll/reorder, register softmax.
// ---------------------------------------------------------------------------

typedef __attribute__((ext_vector_type(16))) _Float16 v16h;
typedef __attribute__((ext_vector_type(8)))  _Float16 v8h;
typedef __attribute__((ext_vector_type(8)))  float    v8f;

#define WMMA_F16(a, b, c) \
  __builtin_amdgcn_wmma_f32_16x16x32_f16(false, (a), false, (b), (short)0, (c), false, false)

// problem constants
constexpr int Bb   = 4;
constexpr int Tt   = 8;
constexpr int Hh   = 56;
constexpr int Ww   = 56;
constexpr int Cc   = 256;
constexpr int HEADS= 8;
constexpr int HD   = 32;           // head dim
constexpr int BT = 2, BH = 7, BW = 7;
constexpr int VOL  = BT * BH * BW; // 98
constexpr int NT = Tt / BT, NH = Hh / BH, NW = Ww / BW; // 4,8,8
constexpr int NCUB = NT * NH * NW; // 256 cuboids per batch
constexpr int BN   = Bb * NCUB;    // 1024 (batch*cuboid)
constexpr int Mtok = BN * VOL;     // 100352 tokens (divisible by 16)
constexpr int VROW = 112;          // vol padded to 7*16 rows
constexpr int VK   = 128;          // vol padded to 4*32 for PV K-dim
constexpr float NEG_INF = -1.0e9f;
constexpr float QSCALE  = 0.17677669529663687f; // 32^-0.5

// assemble a 16x32 f16 A-fragment (contiguous K order, matches B):
// lane l (row = l&15): K[khalf*16 .. +15] in v[0..15]
__device__ inline v16h load_afrag(const _Float16* p) {
  v8h lo = *(const v8h*)p;
  v8h hi = *(const v8h*)(p + 8);
  v16h a;
#pragma unroll
  for (int i = 0; i < 8; ++i) { a[i] = lo[i]; a[i + 8] = hi[i]; }
  return a;
}

// ---------------------------------------------------------------------------
// K0: cast+transpose weights to N-major f16 (B-fragments become contiguous)
// ---------------------------------------------------------------------------
__global__ void prep_weights(const float* __restrict__ qkvw,
                             const float* __restrict__ projw,
                             _Float16* __restrict__ qkvT,
                             _Float16* __restrict__ projT) {
  int idx = blockIdx.x * blockDim.x + threadIdx.x;
  if (idx < 768 * 256) {
    int n = idx >> 8, k = idx & 255;
    qkvT[idx] = (_Float16)qkvw[k * 768 + n];
  }
  if (idx < 256 * 256) {
    int n = idx >> 8, k = idx & 255;
    projT[idx] = (_Float16)projw[k * 256 + n];
  }
}

// ---------------------------------------------------------------------------
// K1: LayerNorm + cyclic roll(-1,-3,-3) + cuboid reorder, f32 -> f16.
// One wave per destination (cuboid-order) token; shfl_xor reductions.
// ---------------------------------------------------------------------------
__global__ __launch_bounds__(256) void ln_shift_reorder(
    const float* __restrict__ x, const float* __restrict__ gamma,
    const float* __restrict__ beta, _Float16* __restrict__ xr) {
  int m    = (blockIdx.x * blockDim.x + threadIdx.x) >> 5;
  int lane = threadIdx.x & 31;
  if (m >= Mtok) return;

  int b  = m / (NCUB * VOL);
  int r  = m - b * (NCUB * VOL);
  int nc = r / VOL, p = r - nc * VOL;
  int nt = nc / (NH * NW), nh = (nc / NW) % NH, nw = nc % NW;
  int pt = p / (BH * BW), ph = (p / BW) % BH, pw = p % BW;
  int tr = nt * BT + pt, hr = nh * BH + ph, wr = nw * BW + pw;
  // xs[tr] = xn[(tr+shift) mod dim]  (np.roll with negative shift)
  int t = (tr + 1) & (Tt - 1);
  int h = hr + 3; if (h >= Hh) h -= Hh;
  int w = wr + 3; if (w >= Ww) w -= Ww;

  const float* row = x + (size_t)(((b * Tt + t) * Hh + h) * Ww + w) * Cc;
  const float4* rp = (const float4*)row;
  float4 a0 = rp[2 * lane];
  float4 a1 = rp[2 * lane + 1];
  float v[8] = {a0.x, a0.y, a0.z, a0.w, a1.x, a1.y, a1.z, a1.w};

  float s = 0.f, ss = 0.f;
#pragma unroll
  for (int i = 0; i < 8; ++i) { s += v[i]; ss += v[i] * v[i]; }
#pragma unroll
  for (int off = 16; off > 0; off >>= 1) {
    s  += __shfl_xor(s, off);
    ss += __shfl_xor(ss, off);
  }
  float mean = s * (1.0f / Cc);
  float var  = ss * (1.0f / Cc) - mean * mean;
  float rstd = rsqrtf(var + 1e-6f);

  int ci = lane * 8;
  v8h st;
#pragma unroll
  for (int i = 0; i < 8; ++i)
    st[i] = (_Float16)((v[i] - mean) * rstd * gamma[ci + i] + beta[ci + i]);
  *(v8h*)(xr + (size_t)m * Cc + ci) = st;
}

// ---------------------------------------------------------------------------
// K2: QKV GEMM  (Mtok x 256) @ (256 x 768), WMMA, 4 waves x (16M x 64N)/block.
// Epilogue scatters q (pre-scaled), k row-major [bn][h][112][32] and v
// transposed [bn][h][32][128] for the attention stage.
// ---------------------------------------------------------------------------
__global__ __launch_bounds__(128) void qkv_gemm(
    const _Float16* __restrict__ xr, const _Float16* __restrict__ qkvT,
    _Float16* __restrict__ qh, _Float16* __restrict__ kh,
    _Float16* __restrict__ vth) {
  int wv = threadIdx.x >> 5, lane = threadIdx.x & 31;
  int m0 = blockIdx.x * 64 + wv * 16;
  int n0 = blockIdx.y * 64;
  int col = lane & 15, khalf = lane >> 4;

  v8f acc[4];
#pragma unroll
  for (int t = 0; t < 4; ++t) acc[t] = (v8f){0, 0, 0, 0, 0, 0, 0, 0};

  for (int kc = 0; kc < 256; kc += 32) {
    v16h a = load_afrag(xr + (size_t)(m0 + col) * 256 + kc + khalf * 16);
#pragma unroll
    for (int t = 0; t < 4; ++t) {
      int n = n0 + t * 16 + col;  // B-fragment: lane col, 16 contiguous K halves
      v16h bf = *(const v16h*)(qkvT + (size_t)n * 256 + kc + khalf * 16);
      acc[t] = WMMA_F16(a, bf, acc[t]);
    }
  }

  int rbase = (lane & 16) ? 8 : 0;  // C-layout: VGPR r holds rows r / r+8
#pragma unroll
  for (int t = 0; t < 4; ++t) {
    int gn  = n0 + t * 16 + col;
    int sec = gn >> 8;          // 0=q 1=k 2=v
    int c   = gn & 255;
    int hh  = c >> 5, d = c & 31;
#pragma unroll
    for (int rr = 0; rr < 8; ++rr) {
      int   mi = m0 + rbase + rr;
      int   bn = mi / VOL, i = mi - bn * VOL;
      float val = acc[t][rr];
      size_t hb = ((size_t)bn * HEADS + hh);
      if (sec == 0)
        qh[(hb * VROW + i) * HD + d] = (_Float16)(val * QSCALE);
      else if (sec == 1)
        kh[(hb * VROW + i) * HD + d] = (_Float16)val;
      else
        vth[(hb * HD + d) * VK + i] = (_Float16)val;
    }
  }
}

// ---------------------------------------------------------------------------
// K3: windowed attention. One 224-thread block (7 waves) per (bn, head).
// Wave w: score rows [16w,16w+16). S = QK^T via WMMA (7 tiles), bias+mask+
// softmax fully in registers (shfl_xor over the 16-lane column group),
// P -> LDS f16 (zero-padded to 112x128), O = P V via WMMA (2x4 tiles).
// ---------------------------------------------------------------------------
__global__ __launch_bounds__(224) void attention(
    const _Float16* __restrict__ qh, const _Float16* __restrict__ kh,
    const _Float16* __restrict__ vth, const float* __restrict__ rel_table,
    _Float16* __restrict__ oh) {
  __shared__ _Float16 sP[VROW][VK];   // 28 KB
  __shared__ float    sRel[507 * 8];  // 16 KB bias table

  int tid  = threadIdx.x;
  int wv   = tid >> 5, lane = tid & 31;
  int blk  = blockIdx.x;
  int bn   = blk >> 3, hh = blk & 7;
  int nc   = bn & (NCUB - 1);

  // zero-pad P and stage the bias table
  for (int idx = tid; idx < (VROW * VK * 2) / 16; idx += blockDim.x)
    ((int4*)sP)[idx] = make_int4(0, 0, 0, 0);
  for (int idx = tid; idx < 507 * 8; idx += blockDim.x)
    sRel[idx] = rel_table[idx];
  __syncthreads();

  int col = lane & 15, khalf = lane >> 4;
  int nt = nc / (NH * NW), nh = (nc / NW) % NH, nw = nc % NW;

  // per-lane column (j) metadata for the 7 score tiles
  int jsm[7], jt_[7], jh_[7], jw_[7];
  bool jok[7];
#pragma unroll
  for (int jt = 0; jt < 7; ++jt) {
    int j = jt * 16 + col;
    jok[jt] = (j < VOL);
    int jj = jok[jt] ? j : VOL - 1;
    int tj = jj / 49, rem = jj % 49, hj = rem / 7, wj = rem % 7;
    jt_[jt] = tj; jh_[jt] = hj; jw_[jt] = wj;
    int tr = nt * BT + tj, hr = nh * BH + hj, wr = nw * BW + wj;
    int rt = tr < Tt - BT ? 0 : (tr < Tt - 1 ? 1 : 2);
    int rh = hr < Hh - BH ? 0 : (hr < Hh - 3 ? 1 : 2);
    int rw = wr < Ww - BW ? 0 : (wr < Ww - 3 ? 1 : 2);
    jsm[jt] = rt * 9 + rh * 3 + rw;
  }

  // ---- S = Q K^T for this wave's 16 rows
  const _Float16* qbase = qh + ((size_t)bn * HEADS + hh) * VROW * HD;
  const _Float16* kbase = kh + ((size_t)bn * HEADS + hh) * VROW * HD;
  float sc[7][8];
  {
    v16h a = load_afrag(qbase + (16 * wv + col) * HD + khalf * 16);
#pragma unroll
    for (int jt = 0; jt < 7; ++jt) {
      v16h bf = *(const v16h*)(kbase + (jt * 16 + col) * HD + khalf * 16);
      v8f c = (v8f){0, 0, 0, 0, 0, 0, 0, 0};
      c = WMMA_F16(a, bf, c);
#pragma unroll
      for (int rr = 0; rr < 8; ++rr) sc[jt][rr] = c[rr];
    }
  }

  // ---- bias + mask + softmax in registers
  int ri0 = 16 * wv + ((lane & 16) ? 8 : 0);
#pragma unroll
  for (int rr = 0; rr < 8; ++rr) {
    int i  = ri0 + rr;
    int ii = (i < VOL) ? i : VOL - 1;         // padded rows: values discarded
    int ti = ii / 49, rem = ii % 49, hi = rem / 7, wi = rem % 7;
    int tr = nt * BT + ti, hr = nh * BH + hi, wr = nw * BW + wi;
    int rt = tr < Tt - BT ? 0 : (tr < Tt - 1 ? 1 : 2);
    int rh = hr < Hh - BH ? 0 : (hr < Hh - 3 ? 1 : 2);
    int rw = wr < Ww - BW ? 0 : (wr < Ww - 3 ? 1 : 2);
    int ism = rt * 9 + rh * 3 + rw;
#pragma unroll
    for (int jt = 0; jt < 7; ++jt) {
      int ridx = (ti - jt_[jt] + 1) * 169 + (hi - jh_[jt] + 6) * 13 +
                 (wi - jw_[jt] + 6);
      float bias = sRel[ridx * 8 + hh];
      sc[jt][rr] = (jok[jt] && ism == jsm[jt]) ? sc[jt][rr] + bias : NEG_INF;
    }
    float mx = -3.0e38f;
#pragma unroll
    for (int jt = 0; jt < 7; ++jt) mx = fmaxf(mx, sc[jt][rr]);
#pragma unroll
    for (int off = 8; off > 0; off >>= 1) mx = fmaxf(mx, __shfl_xor(mx, off));
    float sum = 0.f;
#pragma unroll
    for (int jt = 0; jt < 7; ++jt) {
      float e = __expf(sc[jt][rr] - mx);
      sc[jt][rr] = e;
      sum += e;
    }
#pragma unroll
    for (int off = 8; off > 0; off >>= 1) sum += __shfl_xor(sum, off);
    float inv = 1.0f / sum;
    int ri = ri0 + rr;
#pragma unroll
    for (int jt = 0; jt < 7; ++jt)
      sP[ri][jt * 16 + col] = (_Float16)(sc[jt][rr] * inv);
  }
  __syncthreads();

  // ---- O = P V (K padded to 128, padded P columns are zero)
  const _Float16* vbase = vth + ((size_t)bn * HEADS + hh) * HD * VK;
#pragma unroll
  for (int dt = 0; dt < 2; ++dt) {
    v8f acc = (v8f){0, 0, 0, 0, 0, 0, 0, 0};
#pragma unroll
    for (int jc = 0; jc < 4; ++jc) {
      v16h a  = load_afrag(&sP[16 * wv + col][jc * 32 + khalf * 16]);
      v16h bf = *(const v16h*)(vbase + (dt * 16 + col) * VK + jc * 32 + khalf * 16);
      acc = WMMA_F16(a, bf, acc);
    }
    int rbase = (lane & 16) ? 8 : 0;
#pragma unroll
    for (int rr = 0; rr < 8; ++rr) {
      int ri = 16 * wv + rbase + rr;
      if (ri < VOL) {
        size_t mrow = (size_t)bn * VOL + ri;
        oh[mrow * Cc + hh * HD + dt * 16 + col] = (_Float16)acc[rr];
      }
    }
  }
}

// ---------------------------------------------------------------------------
// K4: output projection (Mtok x 256)@(256 x 256) + bias, WMMA; epilogue
// fuses reverse cuboid reorder + un-roll and scatters f32 rows to d_out.
// ---------------------------------------------------------------------------
__global__ __launch_bounds__(128) void proj_gemm(
    const _Float16* __restrict__ oh, const _Float16* __restrict__ projT,
    const float* __restrict__ pbias, float* __restrict__ out) {
  int wv = threadIdx.x >> 5, lane = threadIdx.x & 31;
  int m0 = blockIdx.x * 64 + wv * 16;
  int n0 = blockIdx.y * 64;
  int col = lane & 15, khalf = lane >> 4;

  v8f acc[4];
#pragma unroll
  for (int t = 0; t < 4; ++t) acc[t] = (v8f){0, 0, 0, 0, 0, 0, 0, 0};

  for (int kc = 0; kc < 256; kc += 32) {
    v16h a = load_afrag(oh + (size_t)(m0 + col) * 256 + kc + khalf * 16);
#pragma unroll
    for (int t = 0; t < 4; ++t) {
      int n = n0 + t * 16 + col;
      v16h bf = *(const v16h*)(projT + (size_t)n * 256 + kc + khalf * 16);
      acc[t] = WMMA_F16(a, bf, acc[t]);
    }
  }

  int rbase = (lane & 16) ? 8 : 0;
#pragma unroll
  for (int rr = 0; rr < 8; ++rr) {
    int mi  = m0 + rbase + rr;
    int bnc = mi / VOL, i = mi - bnc * VOL;
    int b  = bnc >> 8, nc = bnc & (NCUB - 1);
    int nt = nc / (NH * NW), nh = (nc / NW) % NH, nw = nc % NW;
    int pt = i / 49, rem = i % 49, ph = rem / 7, pw = rem % 7;
    int tr = nt * BT + pt, hr = nh * BH + ph, wr = nw * BW + pw;
    int t_ = (tr + 1) & (Tt - 1);          // roll back by +SHIFT
    int h_ = hr + 3; if (h_ >= Hh) h_ -= Hh;
    int w_ = wr + 3; if (w_ >= Ww) w_ -= Ww;
    size_t dst = (size_t)(((b * Tt + t_) * Hh + h_) * Ww + w_) * Cc;
#pragma unroll
    for (int t = 0; t < 4; ++t) {
      int gn = n0 + t * 16 + col;
      out[dst + gn] = acc[t][rr] + pbias[gn];
    }
  }
}

// ---------------------------------------------------------------------------
extern "C" void kernel_launch(void* const* d_in, const int* in_sizes, int n_in,
                              void* d_out, int out_size, void* d_ws,
                              size_t ws_size, hipStream_t stream) {
  const float* x        = (const float*)d_in[0];
  const float* ln_scale = (const float*)d_in[1];
  const float* ln_bias  = (const float*)d_in[2];
  const float* qkv_w    = (const float*)d_in[3];
  const float* rel_tab  = (const float*)d_in[4];
  const float* proj_w   = (const float*)d_in[5];
  const float* proj_b   = (const float*)d_in[6];
  float* out = (float*)d_out;

  // workspace carve-out (256B aligned)
  char* ws = (char*)d_ws;
  size_t off = 0;
  auto carve = [&](size_t bytes) -> void* {
    off = (off + 255) & ~(size_t)255;
    void* p = ws + off;
    off += bytes;
    return p;
  };
  _Float16* xr    = (_Float16*)carve((size_t)Mtok * Cc * 2);          // 51.4 MB
  _Float16* qkvT  = (_Float16*)carve((size_t)768 * 256 * 2);
  _Float16* projT = (_Float16*)carve((size_t)256 * 256 * 2);
  _Float16* qh    = (_Float16*)carve((size_t)BN * HEADS * VROW * HD * 2);
  _Float16* kh    = (_Float16*)carve((size_t)BN * HEADS * VROW * HD * 2);
  _Float16* vth   = (_Float16*)carve((size_t)BN * HEADS * HD * VK * 2);
  _Float16* oh    = (_Float16*)carve((size_t)Mtok * Cc * 2);
  (void)ws_size; (void)in_sizes; (void)n_in; (void)out_size;

  // K0: weights -> N-major f16
  prep_weights<<<(768 * 256 + 255) / 256, 256, 0, stream>>>(qkv_w, proj_w,
                                                            qkvT, projT);
  // K1: LN + roll + reorder (one wave per token)
  ln_shift_reorder<<<(Mtok * 32) / 256, 256, 0, stream>>>(x, ln_scale,
                                                          ln_bias, xr);
  // K2: QKV GEMM (WMMA)
  qkv_gemm<<<dim3(Mtok / 64, 768 / 64), 128, 0, stream>>>(xr, qkvT, qh, kh,
                                                          vth);
  // K3: windowed attention (WMMA), one block per (bn, head)
  attention<<<BN * HEADS, 224, 0, stream>>>(qh, kh, vth, rel_tab, oh);
  // K4: output projection (WMMA) + bias + reverse reorder/unroll
  proj_gemm<<<dim3(Mtok / 64, 256 / 64), 128, 0, stream>>>(oh, projT, proj_b,
                                                           out);
}
